// SelfAttentionBlock_7533372637376
// MI455X (gfx1250) — matrix operand order repaired and checked by reading, not compiled, not run on hardware
//
#include <hip/hip_runtime.h>
#include <hip/hip_bf16.h>

typedef __bf16 bf16_t;
typedef __attribute__((ext_vector_type(16))) __bf16 v16bf;
typedef __attribute__((ext_vector_type(8)))  __bf16 v8bf;
typedef __attribute__((ext_vector_type(4)))  __bf16 v4bf;
typedef __attribute__((ext_vector_type(8)))  float  v8f;

#define D_MODEL 256
#define THREE_D 768
#define NHEAD   8
#define HDIM    32
#define MAXLEN  1024

static __device__ __forceinline__ v8f wmma_bf16(v16bf a, v16bf b, v8f c) {
  // D = A(16x32 bf16) x B(32x16 bf16) + C(16x16 f32)
  return __builtin_amdgcn_wmma_f32_16x16x32_bf16(false, a, false, b, (short)0, c, false, false);
}

static __device__ __forceinline__ v16bf cat8(v8bf lo, v8bf hi) {
  return __builtin_shufflevector(lo, hi, 0,1,2,3,4,5,6,7,8,9,10,11,12,13,14,15);
}

// ---------------- LayerNorm (fp32 in -> bf16 out), one wave per token ----------------
__global__ void k_layernorm(const float* __restrict__ x, const float* __restrict__ g,
                            const float* __restrict__ be, bf16_t* __restrict__ h, int T) {
  int tok  = blockIdx.x * 8 + (threadIdx.x >> 5);
  int lane = threadIdx.x & 31;
  if (tok >= T) return;
  const float4* row = (const float4*)(x + (size_t)tok * D_MODEL);
  float4 xa = row[lane];
  float4 xb = row[lane + 32];
  float s  = xa.x + xa.y + xa.z + xa.w + xb.x + xb.y + xb.z + xb.w;
  float sq = xa.x*xa.x + xa.y*xa.y + xa.z*xa.z + xa.w*xa.w
           + xb.x*xb.x + xb.y*xb.y + xb.z*xb.z + xb.w*xb.w;
  for (int m = 1; m < 32; m <<= 1) { s += __shfl_xor(s, m, 32); sq += __shfl_xor(sq, m, 32); }
  float mu  = s * (1.0f / 256.0f);
  float var = sq * (1.0f / 256.0f) - mu * mu;
  float rs  = rsqrtf(var + 1e-5f);
  const float4* gg = (const float4*)g;
  const float4* bb = (const float4*)be;
  float4 g0 = gg[lane], g1 = gg[lane + 32];
  float4 b0 = bb[lane], b1 = bb[lane + 32];
  v4bf o0, o1;
  o0[0] = (bf16_t)((xa.x - mu) * rs * g0.x + b0.x);
  o0[1] = (bf16_t)((xa.y - mu) * rs * g0.y + b0.y);
  o0[2] = (bf16_t)((xa.z - mu) * rs * g0.z + b0.z);
  o0[3] = (bf16_t)((xa.w - mu) * rs * g0.w + b0.w);
  o1[0] = (bf16_t)((xb.x - mu) * rs * g1.x + b1.x);
  o1[1] = (bf16_t)((xb.y - mu) * rs * g1.y + b1.y);
  o1[2] = (bf16_t)((xb.z - mu) * rs * g1.z + b1.z);
  o1[3] = (bf16_t)((xb.w - mu) * rs * g1.w + b1.w);
  v4bf* out = (v4bf*)(h + (size_t)tok * D_MODEL);
  out[lane]      = o0;
  out[lane + 32] = o1;
}

// ---------------- Weight transpose + fp32->bf16: W[R][C] -> Wt[C][R] ----------------
__global__ void k_cvt_transpose(const float* __restrict__ W, bf16_t* __restrict__ Wt,
                                int R, int C) {
  int i = blockIdx.x * blockDim.x + threadIdx.x;
  if (i >= R * C) return;
  int r = i / C, c = i - r * C;
  Wt[(size_t)c * R + r] = (bf16_t)W[i];
}

// ---------------- QKV GEMM: qkv[T,768] = h[T,256] x Wt[256,768] + b, bf16 out -------
// One wave computes a 16(rows) x 64(cols) tile. grid(mtiles, 3), block 128 (4 waves).
__global__ void k_gemm_qkv(const bf16_t* __restrict__ h, const bf16_t* __restrict__ Wt,
                           const float* __restrict__ bias, bf16_t* __restrict__ qkv, int T) {
  int wave = threadIdx.x >> 5;
  int lane = threadIdx.x & 31;
  int mt   = blockIdx.x;
  int nt64 = blockIdx.y * 4 + wave;          // 0..11, covers 768 cols
  int rowm = mt * 16 + (lane & 15);
  int rclamp = rowm < T ? rowm : T - 1;
  int kb = (lane >> 4) << 3;                 // 0 or 8: A element K-base per lane half
  v8f acc0 = {}, acc1 = {}, acc2 = {}, acc3 = {};
  const bf16_t* arow = h + (size_t)rclamp * D_MODEL;
  for (int k0 = 0; k0 < D_MODEL; k0 += 32) {
    const v8bf* pa = (const v8bf*)(arow + k0 + kb);
    v16bf A = cat8(pa[0], pa[2]);            // elems 0..7:K kb..kb+7, 8..15:K kb+16..kb+23
    const bf16_t* bcol = Wt + (size_t)(k0 + kb) * THREE_D + nt64 * 64 + (lane & 15);  // lane = N col
    v16bf B0, B1, B2, B3;
    for (int s = 0; s < 16; ++s) {
      size_t ko = (size_t)((s & 7) + ((s >> 3) << 4)) * THREE_D;
      B0[s] = bcol[ko]; B1[s] = bcol[ko + 16]; B2[s] = bcol[ko + 32]; B3[s] = bcol[ko + 48];
    }
    acc0 = wmma_bf16(A, B0, acc0);
    acc1 = wmma_bf16(A, B1, acc1);
    acc2 = wmma_bf16(A, B2, acc2);
    acc3 = wmma_bf16(A, B3, acc3);
  }
  int cb = nt64 * 64 + (lane & 15);
  float bj0 = bias[cb], bj1 = bias[cb + 16], bj2 = bias[cb + 32], bj3 = bias[cb + 48];
  int rbase = mt * 16 + kb;                  // C elem e -> row (lane/16)*8 + e
  for (int e = 0; e < 8; ++e) {
    int r = rbase + e;
    if (r < T) {
      bf16_t* out = qkv + (size_t)r * THREE_D + cb;
      out[0]  = (bf16_t)(acc0[e] + bj0);
      out[16] = (bf16_t)(acc1[e] + bj1);
      out[32] = (bf16_t)(acc2[e] + bj2);
      out[48] = (bf16_t)(acc3[e] + bj3);
    }
  }
}

// ---------------- Flash attention: one wave = one (batch, head, 16-row q-tile) ------
// Computes S^T = K * Q^T so the softmax'd P lands directly in A-fragment layout.
__global__ void k_attn(const bf16_t* __restrict__ qkv, const int* __restrict__ offs,
                       bf16_t* __restrict__ o) {
  int bb   = blockIdx.z;
  int hd   = blockIdx.y;
  int wave = threadIdx.x >> 5;
  int lane = threadIdx.x & 31;
  int off0 = offs[bb];
  int len  = offs[bb + 1] - off0;
  int q0   = (blockIdx.x * 4 + wave) * 16;
  if (q0 >= len) return;

  const int qcol = hd * HDIM;
  const int kcol = D_MODEL + hd * HDIM;
  const int vcol = 2 * D_MODEL + hd * HDIM;

  // B fragment = Q^T (32 dims x 16 q): lane = q col, elements = dims. Loaded once.
  v16bf Bq;
  {
    int qq = q0 + (lane & 15); if (qq > len - 1) qq = len - 1;
    const v8bf* pq = (const v8bf*)(qkv + (size_t)(off0 + qq) * THREE_D + qcol + ((lane >> 4) << 3));
    Bq = cat8(pq[0], pq[2]);
  }

  int kb = (lane >> 4) << 3;   // element-row base for C fragments / A element K-base
  v8f O0 = {}, O1 = {};        // O cols 0..15 / 16..31; lane = dim, elems = q rows
  float mrun = -1e30f, lrun = 0.0f;   // per-lane softmax state, q = lane&15
  const float scale = 0.17677669529663687f;  // 1/sqrt(32)

  for (int kt = 0; kt < len; kt += 32) {
    // A fragments = K rows (keys x dims): per-lane contiguous loads
    int ka0 = kt + (lane & 15);       if (ka0 > len - 1) ka0 = len - 1;
    int ka1 = kt + 16 + (lane & 15);  if (ka1 > len - 1) ka1 = len - 1;
    const v8bf* pk0 = (const v8bf*)(qkv + (size_t)(off0 + ka0) * THREE_D + kcol + kb);
    const v8bf* pk1 = (const v8bf*)(qkv + (size_t)(off0 + ka1) * THREE_D + kcol + kb);
    v16bf A0 = cat8(pk0[0], pk0[2]);
    v16bf A1 = cat8(pk1[0], pk1[2]);
    v8f S0 = {}, S1 = {};
    S0 = wmma_bf16(A0, Bq, S0);   // S^T[keys kt+0..15][q]
    S1 = wmma_bf16(A1, Bq, S1);   // S^T[keys kt+16..31][q]

    float s0[8], s1[8];
    float mx = -1e30f;
    for (int e = 0; e < 8; ++e) {
      int key0 = kt + kb + e;
      int key1 = key0 + 16;
      s0[e] = (key0 < len) ? S0[e] * scale : -1e30f;
      s1[e] = (key1 < len) ? S1[e] * scale : -1e30f;
      mx = fmaxf(mx, fmaxf(s0[e], s1[e]));
    }
    mx = fmaxf(mx, __shfl_xor(mx, 16, 32));        // combine the two key halves
    float mnew  = fmaxf(mrun, mx);
    float alpha = __expf(mrun - mnew);
    mrun = mnew;

    float rs = 0.0f;
    v16bf P;   // A-fragment of P: lane = q, elements match key layout exactly
    for (int e = 0; e < 8; ++e) {
      float e0 = __expf(s0[e] - mnew);
      float e1 = __expf(s1[e] - mnew);
      rs += e0 + e1;
      P[e]     = (bf16_t)e0;
      P[e + 8] = (bf16_t)e1;
    }
    rs += __shfl_xor(rs, 16, 32);
    lrun = lrun * alpha + rs;

    // Rescale O accumulators: O elem e belongs to q-row kb+e; alpha lives in lane q.
    for (int e = 0; e < 8; ++e) {
      float aq = __shfl(alpha, kb + e, 32);
      O0[e] *= aq;
      O1[e] *= aq;
    }

    // B fragments = V tile (32 keys x 16 dims): lane = dim, elements = keys.
    v16bf V0, V1;
    for (int s = 0; s < 16; ++s) {
      int kv = kt + kb + (s & 7) + ((s >> 3) << 4); if (kv > len - 1) kv = len - 1;
      const bf16_t* pv = qkv + (size_t)(off0 + kv) * THREE_D + vcol + (lane & 15);
      V0[s] = pv[0];
      V1[s] = pv[16];
    }
    O0 = wmma_bf16(P, V0, O0);
    O1 = wmma_bf16(P, V1, O1);
  }

  // Normalize and store bf16 attention output.
  int dcol = hd * HDIM + (lane & 15);
  for (int e = 0; e < 8; ++e) {
    int qrow = kb + e;
    float linv = 1.0f / __shfl(lrun, qrow, 32);
    int qq = q0 + qrow;
    if (qq < len) {
      bf16_t* orow = o + (size_t)(off0 + qq) * D_MODEL + dcol;
      orow[0]  = (bf16_t)(O0[e] * linv);
      orow[16] = (bf16_t)(O1[e] * linv);
    }
  }
}

// ---------------- Out projection + residual: y = x + o x Wt + b, fp32 out ----------
// One wave per 16x64 tile; 4 waves/block cover all 256 cols. grid(mtiles), block 128.
__global__ void k_outproj(const bf16_t* __restrict__ ob, const bf16_t* __restrict__ Wt,
                          const float* __restrict__ bias, const float* __restrict__ x,
                          float* __restrict__ y, int T) {
  int wave = threadIdx.x >> 5;
  int lane = threadIdx.x & 31;
  int mt   = blockIdx.x;
  int nt64 = wave;
  int rowm = mt * 16 + (lane & 15);
  int rclamp = rowm < T ? rowm : T - 1;
  int kb = (lane >> 4) << 3;
  v8f acc0 = {}, acc1 = {}, acc2 = {}, acc3 = {};
  const bf16_t* arow = ob + (size_t)rclamp * D_MODEL;
  for (int k0 = 0; k0 < D_MODEL; k0 += 32) {
    const v8bf* pa = (const v8bf*)(arow + k0 + kb);
    v16bf A = cat8(pa[0], pa[2]);
    const bf16_t* bcol = Wt + (size_t)(k0 + kb) * D_MODEL + nt64 * 64 + (lane & 15);
    v16bf B0, B1, B2, B3;
    for (int s = 0; s < 16; ++s) {
      size_t ko = (size_t)((s & 7) + ((s >> 3) << 4)) * D_MODEL;
      B0[s] = bcol[ko]; B1[s] = bcol[ko + 16]; B2[s] = bcol[ko + 32]; B3[s] = bcol[ko + 48];
    }
    acc0 = wmma_bf16(A, B0, acc0);
    acc1 = wmma_bf16(A, B1, acc1);
    acc2 = wmma_bf16(A, B2, acc2);
    acc3 = wmma_bf16(A, B3, acc3);
  }
  int cb = nt64 * 64 + (lane & 15);
  float bj0 = bias[cb], bj1 = bias[cb + 16], bj2 = bias[cb + 32], bj3 = bias[cb + 48];
  for (int e = 0; e < 8; ++e) {
    int r = mt * 16 + kb + e;
    if (r < T) {
      const float* xr = x + (size_t)r * D_MODEL + cb;
      float*       yr = y + (size_t)r * D_MODEL + cb;
      yr[0]  = xr[0]  + acc0[e] + bj0;
      yr[16] = xr[16] + acc1[e] + bj1;
      yr[32] = xr[32] + acc2[e] + bj2;
      yr[48] = xr[48] + acc3[e] + bj3;
    }
  }
}

extern "C" void kernel_launch(void* const* d_in, const int* in_sizes, int n_in,
                              void* d_out, int out_size, void* d_ws, size_t ws_size,
                              hipStream_t stream) {
  const float* x    = (const float*)d_in[0];
  const float* win  = (const float*)d_in[1];
  const float* bin  = (const float*)d_in[2];
  const float* wout = (const float*)d_in[3];
  const float* bout = (const float*)d_in[4];
  const float* gam  = (const float*)d_in[5];
  const float* bet  = (const float*)d_in[6];
  const int*   offs = (const int*)d_in[7];
  int T  = in_sizes[0] / D_MODEL;      // total ragged tokens
  int Bn = in_sizes[7] - 1;            // number of sequences

  // Workspace carve-out (256B aligned slices)
  char* ws = (char*)d_ws;
  size_t p = 0;
  auto carve = [&](size_t bytes) -> void* {
    p = (p + 255) & ~(size_t)255;
    void* r = ws + p;
    p += bytes;
    return r;
  };
  bf16_t* hbf = (bf16_t*)carve((size_t)T * D_MODEL * sizeof(bf16_t));
  bf16_t* qkv = (bf16_t*)carve((size_t)T * THREE_D * sizeof(bf16_t));
  bf16_t* obf = (bf16_t*)carve((size_t)T * D_MODEL * sizeof(bf16_t));
  bf16_t* WtA = (bf16_t*)carve((size_t)D_MODEL * THREE_D * sizeof(bf16_t));
  bf16_t* WtO = (bf16_t*)carve((size_t)D_MODEL * D_MODEL * sizeof(bf16_t));
  (void)ws_size; (void)n_in; (void)out_size;

  int mtiles = (T + 15) / 16;

  k_layernorm<<<dim3((T + 7) / 8), dim3(256), 0, stream>>>(x, gam, bet, hbf, T);
  k_cvt_transpose<<<dim3((THREE_D * D_MODEL + 255) / 256), dim3(256), 0, stream>>>(
      win, WtA, THREE_D, D_MODEL);
  k_cvt_transpose<<<dim3((D_MODEL * D_MODEL + 255) / 256), dim3(256), 0, stream>>>(
      wout, WtO, D_MODEL, D_MODEL);
  k_gemm_qkv<<<dim3(mtiles, 3), dim3(128), 0, stream>>>(hbf, WtA, bin, qkv, T);
  k_attn<<<dim3(MAXLEN / 64, NHEAD, Bn), dim3(128), 0, stream>>>(qkv, offs, obf);
  k_outproj<<<dim3(mtiles), dim3(128), 0, stream>>>(obf, WtO, bout, x, (float*)d_out, T);
}
